// Decoder_68564857913413
// MI455X (gfx1250) — matrix operand order repaired and checked by reading, not compiled, not run on hardware
//
#include <hip/hip_runtime.h>

typedef __bf16 bf16_t;
typedef __attribute__((ext_vector_type(16))) __bf16 v16bf;
typedef __attribute__((ext_vector_type(8)))  float  v8f;
typedef __attribute__((ext_vector_type(4)))  int    v4i;

#define B_    64
#define L_    256
#define ENC_  512
#define F_    800
#define MEL_  80
#define DEC_  1024
#define PRE_  256
#define ATT_  128
#define FILT_ 32
#define KCV_  31
#define NG_   4096           // 4*DEC
#define KA_   1792           // PRE + ENC + DEC  (attention LSTM K)
#define KG_   2560           // DEC + ENC + DEC  (generator LSTM K)
#define KH_   1536           // DEC + ENC        (head K)
#define ZH_   0.1f
#define ZC_   0.1f

#if defined(__gfx1250__) && __has_builtin(__builtin_amdgcn_global_load_async_to_lds_b128)
#define HAVE_ASYNC_LDS 1
#else
#define HAVE_ASYNC_LDS 0
#endif

// Typed address-space casts for the async global->LDS builtin:
// param0 is int4 in AS(1) (prints as "int __vector(4) __device__ *"),
// param1 is the LDS-side pointer in AS(3).
#define ASYNC_GPTR(p) ((__attribute__((address_space(1))) v4i*)(p))
#define ASYNC_LPTR(p) ((__attribute__((address_space(3))) v4i*)(p))

// ---------------------------------------------------------------------------
// WMMA helpers (bf16 A/B, f32 accumulate), 16x16x32; A/B use this code's own k order (not the ISA table order):
// A (16x32, row-major): lane m = lane&15, one contiguous 16-elem chunk at
//   k = 16*(lane>>4).
// B is pre-packed so each lane's 16 values are contiguous:
//   packed[((k>>5)*N + n)*32 + (k&31)], lane reads offset 16*(lane>>4).
// C/D (f32): lane n = lane&15, vgpr r -> m = r + 8*(lane>>4).
// ---------------------------------------------------------------------------
__device__ inline v8f wmma_bf16(v16bf a, v16bf b, v8f c) {
  return __builtin_amdgcn_wmma_f32_16x16x32_bf16(
      /*neg_a=*/false, a, /*neg_b=*/false, b,
      /*c_mod=*/(short)0, c, /*reuse_a=*/false, /*reuse_b=*/false);
}

__device__ inline v16bf load_a(const bf16_t* A, long lda, int m0, int k0) {
  int lane = threadIdx.x & 31;
  const bf16_t* row = A + (long)(m0 + (lane & 15)) * lda + k0;
  int kb = (lane >> 4) * 16;
  v16bf a;
#pragma unroll
  for (int e = 0; e < 16; ++e) a[e] = row[kb + e];
  return a;
}

__device__ inline v16bf load_b(const bf16_t* Bp, int N, int k0, int n0) {
  int lane = threadIdx.x & 31;
  const bf16_t* p = Bp + ((long)(k0 >> 5) * N + (n0 + (lane & 15))) * 32 + (lane >> 4) * 16;
  v16bf b;
#pragma unroll
  for (int e = 0; e < 16; ++e) b[e] = p[e];
  return b;
}

// Read a B fragment for gate-tile g from an LDS staging buffer (4 tiles of
// 512 bf16 each, tile-contiguous exactly like the packed global layout).
__device__ inline v16bf load_b_lds(const bf16_t* base, int g) {
  int lane = threadIdx.x & 31;
  const bf16_t* p = base + g * 512 + (lane & 15) * 32 + (lane >> 4) * 16;
  v16bf b;
#pragma unroll
  for (int e = 0; e < 16; ++e) b[e] = p[e];
  return b;
}

__device__ inline float sigf(float x) { return 1.f / (1.f + expf(-x)); }

// Stage the 4 gate B-tiles (4 KB) for k-chunk k0 into LDS. 128 threads:
// wave g copies its own gate tile (32 lanes x 32 B). In the packed layout a
// 16x32 tile is 512 contiguous bf16.
__device__ inline void stage_b(const bf16_t* __restrict__ W, int n0, int k0,
                               bf16_t* __restrict__ dst) {
  int g  = threadIdx.x >> 5;
  int ln = threadIdx.x & 31;
  const bf16_t* src = W + ((long)(k0 >> 5) * NG_ + n0 + g * DEC_) * 32 + ln * 16;
  bf16_t* d = dst + g * 512 + ln * 16;
#if HAVE_ASYNC_LDS
  __builtin_amdgcn_global_load_async_to_lds_b128(ASYNC_GPTR(src), ASYNC_LPTR(d), 0, 0);
  __builtin_amdgcn_global_load_async_to_lds_b128(ASYNC_GPTR(src + 8), ASYNC_LPTR(d + 8), 0, 0);
#else
  const uint4* s4 = (const uint4*)src;
  uint4* d4 = (uint4*)d;
  d4[0] = s4[0];
  d4[1] = s4[1];
#endif
}

__device__ inline void wait_stage_and_barrier() {
#if HAVE_ASYNC_LDS
#if __has_builtin(__builtin_amdgcn_s_wait_asynccnt)
  __builtin_amdgcn_s_wait_asynccnt(0);
#else
  asm volatile("s_wait_asynccnt 0x0" ::: "memory");
#endif
#endif
  __syncthreads();
}

// ---------------------------------------------------------------------------
// Setup kernels
// ---------------------------------------------------------------------------

// src is (N x Ksrc) row-major f32; write bf16 packed B-operand layout,
// rows [koff, koff+Kloc) of a (Ktot x Npack) matrix; zero-pad k >= Ksrc.
__global__ void pack_weight_kernel(const float* __restrict__ src, int N, int Ksrc,
                                   int Kloc, int koff, bf16_t* __restrict__ dst, int Npack) {
  long i = (long)blockIdx.x * blockDim.x + threadIdx.x;
  long total = (long)Kloc * N;
  if (i >= total) return;
  int n  = (int)(i % N);
  int kl = (int)(i / N);
  float v = (kl < Ksrc) ? src[(long)n * Ksrc + kl] : 0.f;
  int kg = koff + kl;
  dst[((long)(kg >> 5) * Npack + n) * 32 + (kg & 31)] = (bf16_t)v;
}

__global__ void bias_add_kernel(const float* a, const float* b, float* o, int n) {
  int i = blockIdx.x * blockDim.x + threadIdx.x;
  if (i < n) o[i] = a[i] + b[i];
}

__global__ void f32_to_bf16_kernel(const float* s, bf16_t* d, long n) {
  long i = (long)blockIdx.x * blockDim.x + threadIdx.x;
  if (i < n) d[i] = (bf16_t)s[i];
}

__global__ void zero_ws_kernel(unsigned int* p, long nwords) {
  long i = (long)blockIdx.x * blockDim.x + threadIdx.x;
  if (i < nwords) p[i] = 0u;
}

// tgt_bf (B, F, 96): frame 0 = zeros, frame t = target[b, :, t-1]; cols 80..95 zero.
__global__ void build_tgt_kernel(const float* __restrict__ target, bf16_t* __restrict__ tgt) {
  long i = (long)blockIdx.x * blockDim.x + threadIdx.x;
  long total = (long)B_ * F_ * 96;
  if (i >= total) return;
  int k = (int)(i % 96);
  long bt = i / 96;
  int t = (int)(bt % F_);
  int b = (int)(bt / F_);
  float v = 0.f;
  if (t > 0 && k < MEL_) v = target[((long)b * MEL_ + k) * F_ + (t - 1)];
  tgt[i] = (bf16_t)v;
}

// Generic GEMM: C(MxN) = A(MxK, bf16 row-major) * Bpacked(KxN bf16) [+bias][relu]
// block=128 (4 waves), each wave owns one 16x16 output tile.
__global__ __launch_bounds__(128) void gemm_kernel(
    const bf16_t* __restrict__ A, long lda, const bf16_t* __restrict__ Bp,
    int M, int N, int Ktot, const float* __restrict__ bias, int relu,
    float* __restrict__ Cf, bf16_t* __restrict__ Cb, long ldc) {
  int tiles_n = N >> 4;
  int wg = blockIdx.x * 4 + (threadIdx.x >> 5);
  int total = (M >> 4) * tiles_n;
  if (wg >= total) return;                 // wave-uniform: EXEC stays full
  int m0 = (wg / tiles_n) << 4;
  int n0 = (wg % tiles_n) << 4;
  v8f acc = {};
  for (int k0 = 0; k0 < Ktot; k0 += 32) {
    v16bf a = load_a(A, lda, m0, k0);
    v16bf b = load_b(Bp, N, k0, n0);
    acc = wmma_bf16(a, b, acc);
  }
  int lane = threadIdx.x & 31;
  int n = n0 + (lane & 15);
  float bb = bias ? bias[n] : 0.f;
#pragma unroll
  for (int r = 0; r < 8; ++r) {
    int m = m0 + r + 8 * (lane >> 4);
    float v = acc[r] + bb;
    if (relu) v = v > 0.f ? v : 0.f;
    if (Cf) Cf[(long)m * ldc + n] = v;
    if (Cb) Cb[(long)m * ldc + n] = (bf16_t)v;
  }
}

// ---------------------------------------------------------------------------
// Fused LSTM step: gates = X*Wcat + bias, then sigmoid/tanh + zoneout blend.
// X is the concat of up to 3 bf16 segments: [0,k1)=X0, [k1,k2)=X1, [k2,ktot)=X2.
// grid.x = DEC/16 = 64 (one block per 16 hidden units), block = 128 (4 waves,
// one per 16-row batch tile). All 4 waves consume the SAME 4 gate B-tiles, so
// they are staged once per k-step into double-buffered LDS via the async
// global->LDS path (ASYNCcnt) and shared; one barrier per k-iteration.
// ---------------------------------------------------------------------------
__global__ __launch_bounds__(128) void lstm_step_kernel(
    const bf16_t* __restrict__ X0, long s0, int k1,
    const bf16_t* __restrict__ X1, long s1, int k2,
    const bf16_t* __restrict__ X2, long s2, int ktot,
    const bf16_t* __restrict__ W, const float* __restrict__ bias,
    float* __restrict__ h, float* __restrict__ c, bf16_t* __restrict__ h_bf) {
  __shared__ __align__(16) bf16_t sB[2][4 * 512];   // 2 x 4 KB
  int wave = threadIdx.x >> 5;
  int lane = threadIdx.x & 31;
  int m0 = wave << 4;
  int n0 = blockIdx.x << 4;
  v8f acc0 = {}, acc1 = {}, acc2 = {}, acc3 = {};
  stage_b(W, n0, 0, sB[0]);
  for (int k0 = 0; k0 < ktot; k0 += 32) {
    bf16_t* bufB = sB[(k0 >> 5) & 1];
    // stage issued previous iteration is complete & visible after this:
    wait_stage_and_barrier();
    if (k0 + 32 < ktot) stage_b(W, n0, k0 + 32, sB[((k0 >> 5) + 1) & 1]);
    const bf16_t* p; long st; int kk;
    if (k0 >= k2)      { p = X2; st = s2; kk = k0 - k2; }
    else if (k0 >= k1) { p = X1; st = s1; kk = k0 - k1; }
    else               { p = X0; st = s0; kk = k0; }
    v16bf a = load_a(p, st, m0, kk);
    v16bf b0 = load_b_lds(bufB, 0);
    v16bf b1 = load_b_lds(bufB, 1);
    v16bf b2 = load_b_lds(bufB, 2);
    v16bf b3 = load_b_lds(bufB, 3);
    acc0 = wmma_bf16(a, b0, acc0);
    acc1 = wmma_bf16(a, b1, acc1);
    acc2 = wmma_bf16(a, b2, acc2);
    acc3 = wmma_bf16(a, b3, acc3);
  }
  int n = n0 + (lane & 15);
  float bi = bias[n], bf = bias[n + DEC_], bg = bias[n + 2 * DEC_], bo = bias[n + 3 * DEC_];
#pragma unroll
  for (int r = 0; r < 8; ++r) {
    int m = m0 + r + 8 * (lane >> 4);
    long idx = (long)m * DEC_ + n;
    float iv = acc0[r] + bi, fv = acc1[r] + bf, gv = acc2[r] + bg, ov = acc3[r] + bo;
    float c_old = c[idx], h_old = h[idx];
    float nc = sigf(fv) * c_old + sigf(iv) * tanhf(gv);
    float nh = sigf(ov) * tanhf(nc);
    float hn = ZH_ * h_old + (1.f - ZH_) * nh;
    float cn = ZC_ * c_old + (1.f - ZC_) * nc;
    h[idx] = hn; c[idx] = cn; h_bf[idx] = (bf16_t)hn;
  }
}

// ---------------------------------------------------------------------------
// Location-sensitive attention: one block per batch element, 256 threads
// (one per encoder position). conv -> energies -> softmax -> context.
// ---------------------------------------------------------------------------
__global__ __launch_bounds__(256) void attention_kernel(
    const float* __restrict__ q,       // (B, ATT)
    const float* __restrict__ mem_t,   // (B*L, ATT)
    const float* __restrict__ enc,     // (B, L, ENC) f32
    const unsigned char* __restrict__ mask,  // (B, L)
    const float* __restrict__ conv_w,  // (FILT, 1, KCV)
    const float* __restrict__ W_loc,   // (ATT, FILT)
    const float* __restrict__ v_att,   // (ATT)
    float* __restrict__ cum,           // (B, L)
    float* __restrict__ ctx_f, bf16_t* __restrict__ ctx_b,
    float* __restrict__ out_align, int t) {
  __shared__ float s_cum[L_ + 2 * (KCV_ / 2)];
  __shared__ float s_w[L_];
  __shared__ float s_red[256];
  const int b = blockIdx.x;
  const int l = threadIdx.x;
  const int HP = KCV_ / 2;  // 15
  float cum_old = cum[(long)b * L_ + l];
  s_cum[HP + l] = cum_old;
  if (l < HP) { s_cum[l] = 0.f; s_cum[HP + L_ + l] = 0.f; }
  __syncthreads();
  // conv over cumulative weights (32 filters, 'same' padding)
  float cf[FILT_];
#pragma unroll
  for (int f = 0; f < FILT_; ++f) {
    float s = 0.f;
    const float* wf = conv_w + f * KCV_;
#pragma unroll
    for (int tp = 0; tp < KCV_; ++tp) s += wf[tp] * s_cum[l + tp];
    cf[f] = s;
  }
  // energy: e = sum_a v[a] * tanh(q[a] + mem[a] + sum_f Wloc[a][f]*cf[f])
  const float* mrow = mem_t + ((long)b * L_ + l) * ATT_;
  const float* qrow = q + (long)b * ATT_;
  float e = 0.f;
  for (int a = 0; a < ATT_; ++a) {
    float s = qrow[a] + mrow[a];
    const float* wl = W_loc + a * FILT_;
#pragma unroll
    for (int f = 0; f < FILT_; ++f) s += wl[f] * cf[f];
    e += v_att[a] * tanhf(s);
  }
  if (!mask[(long)b * L_ + l]) e = -1e9f;
  // softmax over L (block reduction)
  s_red[l] = e;
  __syncthreads();
  for (int off = 128; off >= 1; off >>= 1) {
    if (l < off) s_red[l] = fmaxf(s_red[l], s_red[l + off]);
    __syncthreads();
  }
  float mx = s_red[0];
  __syncthreads();
  float ex = expf(e - mx);
  s_red[l] = ex;
  __syncthreads();
  for (int off = 128; off >= 1; off >>= 1) {
    if (l < off) s_red[l] += s_red[l + off];
    __syncthreads();
  }
  float w = ex / s_red[0];
  s_w[l] = w;
  float nc = cum_old + w;
  cum[(long)b * L_ + l] = nc;
  out_align[((long)b * F_ + t) * L_ + l] = nc;
  __syncthreads();
  // context = sum_l w[l] * enc[b, l, :]
  for (int e0 = l; e0 < ENC_; e0 += 256) {
    float s = 0.f;
    const float* eb = enc + (long)b * L_ * ENC_ + e0;
    for (int ll = 0; ll < L_; ++ll) s += s_w[ll] * eb[(long)ll * ENC_];
    ctx_f[(long)b * ENC_ + e0] = s;
    ctx_b[(long)b * ENC_ + e0] = (bf16_t)s;
  }
}

// ---------------------------------------------------------------------------
// Output heads: frame = [h_g|ctx] * W_frame^T + b_frame (WMMA, 5 N-tiles),
// stop head as scalar dot (block 5).
// ---------------------------------------------------------------------------
__global__ __launch_bounds__(128) void heads_kernel(
    const bf16_t* __restrict__ Hg, const bf16_t* __restrict__ Ctx,
    const bf16_t* __restrict__ Wh, const float* __restrict__ b_frame,
    const float* __restrict__ h_g_f, const float* __restrict__ ctx_f,
    const float* __restrict__ W_stop, const float* __restrict__ b_stop,
    float* __restrict__ out_spec, float* __restrict__ out_stop, int t) {
  if (blockIdx.x < 5) {
    int wave = threadIdx.x >> 5;
    int lane = threadIdx.x & 31;
    int m0 = wave << 4;
    int n0 = blockIdx.x << 4;
    v8f acc = {};
    for (int k0 = 0; k0 < KH_; k0 += 32) {
      const bf16_t* p; long st; int kk;
      if (k0 >= DEC_) { p = Ctx; st = ENC_; kk = k0 - DEC_; }
      else            { p = Hg;  st = DEC_; kk = k0; }
      v16bf a = load_a(p, st, m0, kk);
      v16bf b = load_b(Wh, MEL_, k0, n0);
      acc = wmma_bf16(a, b, acc);
    }
    int n = n0 + (lane & 15);
    float bb = b_frame[n];
#pragma unroll
    for (int r = 0; r < 8; ++r) {
      int m = m0 + r + 8 * (lane >> 4);
      out_spec[((long)m * F_ + t) * MEL_ + n] = acc[r] + bb;
    }
  } else {
    int b = threadIdx.x;
    if (b < B_) {
      float s = b_stop[0];
      const float* hg = h_g_f + (long)b * DEC_;
      const float* cx = ctx_f + (long)b * ENC_;
      for (int k = 0; k < DEC_; ++k) s += hg[k] * W_stop[k];
      for (int k = 0; k < ENC_; ++k) s += cx[k] * W_stop[DEC_ + k];
      out_stop[(long)b * F_ + t] = s;
    }
  }
}

// ---------------------------------------------------------------------------
// Host launch
// ---------------------------------------------------------------------------
extern "C" void kernel_launch(void* const* d_in, const int* in_sizes, int n_in,
                              void* d_out, int out_size, void* d_ws, size_t ws_size,
                              hipStream_t stream) {
  (void)in_sizes; (void)n_in; (void)out_size; (void)ws_size;
  const float* encoded = (const float*)d_in[0];
  const float* target  = (const float*)d_in[1];
  const unsigned char* mask = (const unsigned char*)d_in[2];
  const float* W_p1 = (const float*)d_in[3];
  const float* b_p1 = (const float*)d_in[4];
  const float* W_p2 = (const float*)d_in[5];
  const float* b_p2 = (const float*)d_in[6];
  const float* W_mem = (const float*)d_in[7];
  const float* W_q = (const float*)d_in[8];
  const float* conv_loc = (const float*)d_in[9];
  const float* W_loc = (const float*)d_in[10];
  const float* v_att = (const float*)d_in[11];
  const float* Wih_a = (const float*)d_in[12];
  const float* Whh_a = (const float*)d_in[13];
  const float* bih_a = (const float*)d_in[14];
  const float* bhh_a = (const float*)d_in[15];
  const float* Wih_g = (const float*)d_in[16];
  const float* Whh_g = (const float*)d_in[17];
  const float* bih_g = (const float*)d_in[18];
  const float* bhh_g = (const float*)d_in[19];
  const float* W_frame = (const float*)d_in[20];
  const float* b_frame = (const float*)d_in[21];
  const float* W_stop = (const float*)d_in[22];
  const float* b_stop = (const float*)d_in[23];

  float* out_spec  = (float*)d_out;                                   // (B,F,MEL)
  float* out_stop  = out_spec + (size_t)B_ * F_ * MEL_;               // (B,F)
  float* out_align = out_stop + (size_t)B_ * F_;                      // (B,F,L)

  // Workspace bump allocator (deterministic layout every call).
  char* ws = (char*)d_ws;
  size_t off = 0;
  auto alloc = [&](size_t bytes) -> char* {
    off = (off + 255) & ~(size_t)255;
    char* p = ws + off;
    off += bytes;
    return p;
  };
  bf16_t* Wcat_a = (bf16_t*)alloc((size_t)KA_ * NG_ * 2);
  bf16_t* Wcat_g = (bf16_t*)alloc((size_t)KG_ * NG_ * 2);
  bf16_t* Wq_t   = (bf16_t*)alloc((size_t)DEC_ * ATT_ * 2);
  bf16_t* Whead  = (bf16_t*)alloc((size_t)KH_ * MEL_ * 2);
  bf16_t* Wmem_t = (bf16_t*)alloc((size_t)ENC_ * ATT_ * 2);
  bf16_t* Wp1_t  = (bf16_t*)alloc((size_t)96 * PRE_ * 2);
  bf16_t* Wp2_t  = (bf16_t*)alloc((size_t)PRE_ * PRE_ * 2);
  float*  bias_a = (float*)alloc((size_t)NG_ * 4);
  float*  bias_g = (float*)alloc((size_t)NG_ * 4);
  float*  mem_t  = (float*)alloc((size_t)B_ * L_ * ATT_ * 4);
  bf16_t* pre_bf = (bf16_t*)alloc((size_t)B_ * F_ * PRE_ * 2);
  size_t st_begin = (off + 255) & ~(size_t)255;
  float*  h_a   = (float*)alloc((size_t)B_ * DEC_ * 4);
  float*  c_a   = (float*)alloc((size_t)B_ * DEC_ * 4);
  float*  h_g   = (float*)alloc((size_t)B_ * DEC_ * 4);
  float*  c_g   = (float*)alloc((size_t)B_ * DEC_ * 4);
  float*  ctx_f = (float*)alloc((size_t)B_ * ENC_ * 4);
  float*  cum   = (float*)alloc((size_t)B_ * L_ * 4);
  float*  qbuf  = (float*)alloc((size_t)B_ * ATT_ * 4);
  bf16_t* h_a_bf[2], *h_g_bf[2];
  h_a_bf[0] = (bf16_t*)alloc((size_t)B_ * DEC_ * 2);
  h_a_bf[1] = (bf16_t*)alloc((size_t)B_ * DEC_ * 2);
  h_g_bf[0] = (bf16_t*)alloc((size_t)B_ * DEC_ * 2);
  h_g_bf[1] = (bf16_t*)alloc((size_t)B_ * DEC_ * 2);
  bf16_t* ctx_bf = (bf16_t*)alloc((size_t)B_ * ENC_ * 2);
  size_t st_end = off;
  // Setup temporaries (reused region; enc_bf aliases tgt/pre1 after prenet)
  bf16_t* tgt_bf  = (bf16_t*)alloc((size_t)B_ * F_ * 96 * 2);
  bf16_t* pre1_bf = (bf16_t*)alloc((size_t)B_ * F_ * PRE_ * 2);
  bf16_t* enc_bf  = tgt_bf;

  auto blocks = [](long total, int bs) { return (unsigned)((total + bs - 1) / bs); };

  // ---- Setup: pack weights to bf16 fragment layout ----
  pack_weight_kernel<<<blocks((long)768 * NG_, 256), 256, 0, stream>>>(Wih_a, NG_, 768, 768, 0, Wcat_a, NG_);
  pack_weight_kernel<<<blocks((long)1024 * NG_, 256), 256, 0, stream>>>(Whh_a, NG_, 1024, 1024, 768, Wcat_a, NG_);
  pack_weight_kernel<<<blocks((long)1536 * NG_, 256), 256, 0, stream>>>(Wih_g, NG_, 1536, 1536, 0, Wcat_g, NG_);
  pack_weight_kernel<<<blocks((long)1024 * NG_, 256), 256, 0, stream>>>(Whh_g, NG_, 1024, 1024, 1536, Wcat_g, NG_);
  pack_weight_kernel<<<blocks((long)DEC_ * ATT_, 256), 256, 0, stream>>>(W_q, ATT_, DEC_, DEC_, 0, Wq_t, ATT_);
  pack_weight_kernel<<<blocks((long)KH_ * MEL_, 256), 256, 0, stream>>>(W_frame, MEL_, KH_, KH_, 0, Whead, MEL_);
  pack_weight_kernel<<<blocks((long)ENC_ * ATT_, 256), 256, 0, stream>>>(W_mem, ATT_, ENC_, ENC_, 0, Wmem_t, ATT_);
  pack_weight_kernel<<<blocks((long)96 * PRE_, 256), 256, 0, stream>>>(W_p1, PRE_, MEL_, 96, 0, Wp1_t, PRE_);
  pack_weight_kernel<<<blocks((long)PRE_ * PRE_, 256), 256, 0, stream>>>(W_p2, PRE_, PRE_, PRE_, 0, Wp2_t, PRE_);
  bias_add_kernel<<<blocks(NG_, 256), 256, 0, stream>>>(bih_a, bhh_a, bias_a, NG_);
  bias_add_kernel<<<blocks(NG_, 256), 256, 0, stream>>>(bih_g, bhh_g, bias_g, NG_);

  // ---- Setup: prenet over all frames (WMMA) ----
  build_tgt_kernel<<<blocks((long)B_ * F_ * 96, 256), 256, 0, stream>>>(target, tgt_bf);
  {
    int M = B_ * F_;
    long tiles = (long)(M / 16) * (PRE_ / 16);
    gemm_kernel<<<blocks(tiles, 4), 128, 0, stream>>>(tgt_bf, 96, Wp1_t, M, PRE_, 96, b_p1, 1, nullptr, pre1_bf, PRE_);
    gemm_kernel<<<blocks(tiles, 4), 128, 0, stream>>>(pre1_bf, PRE_, Wp2_t, M, PRE_, PRE_, b_p2, 1, nullptr, pre_bf, PRE_);
  }
  // ---- Setup: mem_t = encoded @ W_mem^T (WMMA) ----
  f32_to_bf16_kernel<<<blocks((long)B_ * L_ * ENC_, 256), 256, 0, stream>>>(encoded, enc_bf, (long)B_ * L_ * ENC_);
  {
    int M = B_ * L_;
    long tiles = (long)(M / 16) * (ATT_ / 16);
    gemm_kernel<<<blocks(tiles, 4), 128, 0, stream>>>(enc_bf, ENC_, Wmem_t, M, ATT_, ENC_, nullptr, 0, mem_t, nullptr, ATT_);
  }
  // ---- Setup: zero all recurrent state ----
  zero_ws_kernel<<<blocks((long)(st_end - st_begin) / 4, 256), 256, 0, stream>>>(
      (unsigned int*)(ws + st_begin), (long)(st_end - st_begin) / 4);

  // ---- Sequential decoder scan ----
  for (int t = 0; t < F_; ++t) {
    int cur = t & 1, nxt = cur ^ 1;
    // 1. attention LSTM: X = [pre_t(256) | ctx(512) | h_a(1024)]
    lstm_step_kernel<<<DEC_ / 16, 128, 0, stream>>>(
        pre_bf + (long)t * PRE_, (long)F_ * PRE_, PRE_,
        ctx_bf, (long)ENC_, PRE_ + ENC_,
        h_a_bf[cur], (long)DEC_, KA_,
        Wcat_a, bias_a, h_a, c_a, h_a_bf[nxt]);
    // 2. q = h_a_new @ W_q^T  (WMMA, 32 tiles)
    gemm_kernel<<<8, 128, 0, stream>>>(h_a_bf[nxt], DEC_, Wq_t, B_, ATT_, DEC_, nullptr, 0, qbuf, nullptr, ATT_);
    // 3. location-sensitive attention
    attention_kernel<<<B_, 256, 0, stream>>>(qbuf, mem_t, encoded, mask, conv_loc, W_loc, v_att,
                                             cum, ctx_f, ctx_bf, out_align, t);
    // 4. generator LSTM: X = [h_a(1024) | ctx(512) | h_g(1024)]
    lstm_step_kernel<<<DEC_ / 16, 128, 0, stream>>>(
        h_a_bf[nxt], (long)DEC_, DEC_,
        ctx_bf, (long)ENC_, DEC_ + ENC_,
        h_g_bf[cur], (long)DEC_, KG_,
        Wcat_g, bias_g, h_g, c_g, h_g_bf[nxt]);
    // 5. output heads
    heads_kernel<<<6, 128, 0, stream>>>(h_g_bf[nxt], ctx_bf, Whead, b_frame,
                                        h_g, ctx_f, W_stop, b_stop, out_spec, out_stop, t);
  }
}
